// InvariantPointAttention_59476707115300
// MI455X (gfx1250) — compile-verified, matrix operand order checked by reading, not run on hardware
//
#include <hip/hip_runtime.h>
#include <math.h>

#define NN 1024
#define C1 384
#define C2 128
#define HH 12
#define FEAT 2112
#define OUTD 384

typedef __attribute__((ext_vector_type(16))) __bf16 v16bf;
typedef __attribute__((ext_vector_type(8)))  float  v8f;

union BF16x16 { v16bf v; __bf16 e[16]; };

__device__ __forceinline__ v8f wmma_bf16(v16bf a, v16bf b, v8f c) {
  return __builtin_amdgcn_wmma_f32_16x16x32_bf16(false, a, false, b, (short)0, c, false, false);
}
__device__ __forceinline__ v16bf frag_bf16(const __bf16* p) {
  return *(const v16bf*)p;   // 16 contiguous bf16 (32B)
}
__device__ __forceinline__ v16bf frag_cvt(const float* p) {
  BF16x16 u;
#pragma unroll
  for (int i = 0; i < 16; ++i) u.e[i] = (__bf16)p[i];
  return u.v;
}
__device__ __forceinline__ v8f zero8() {
  v8f z;
#pragma unroll
  for (int i = 0; i < 8; ++i) z[i] = 0.f;
  return z;
}
__device__ __forceinline__ float softplusf(float x) {
  float ax = fabsf(x);
  return fmaxf(x, 0.f) + log1pf(__expf(-ax));
}
// gfx1250 async DMA: copy 16B/lane from global directly into LDS (ASYNCcnt).
__device__ __forceinline__ void async_copy16(unsigned lds_off, const void* gptr) {
  asm volatile("global_load_async_to_lds_b128 %0, %1, off"
               :: "v"(lds_off), "v"(gptr) : "memory");
}
__device__ __forceinline__ void wait_async0() {
  asm volatile("s_wait_asynccnt 0" ::: "memory");
}

// ---------------------------------------------------------------------------
// Prep: w_out^T -> bf16 [384][2112], w2d^T -> bf16 [16][128] (rows>=12 zero)
// ---------------------------------------------------------------------------
__global__ __launch_bounds__(256) void ipa_wprep(
    const float* __restrict__ w_out, const float* __restrict__ w2d,
    __bf16* __restrict__ wT, __bf16* __restrict__ w2dT) {
  int id = blockIdx.x * 256 + threadIdx.x;
  if (id < OUTD * FEAT) {
    int nn = id / FEAT, k = id % FEAT;
    wT[id] = (__bf16)w_out[(size_t)k * OUTD + nn];
  } else {
    int j = id - OUTD * FEAT;
    if (j < 16 * C2) {
      int h = j / C2, c = j % C2;
      w2dT[j] = (h < HH) ? (__bf16)w2d[c * HH + h] : (__bf16)0.f;
    }
  }
}

// ---------------------------------------------------------------------------
// Projections + rigid transforms.
//  qcomb/kcomb: bf16 [12][1024][32] = [ q_s*0.25 | w_h*q_pt_global | 0 ]
//  Qn/Kn: f32 [1024][12] = sum_p |pt_global|^2
//  vsT:  bf16 [12][16][1024]   (v_s transposed, k-contiguous)
//  vptT: bf16 [12][32][1024]   (rows 0..23 = p*3+i of v_pt_global, 24..31 = 0)
// ---------------------------------------------------------------------------
__global__ __launch_bounds__(256) void ipa_project(
    const float* __restrict__ in1d, const float* __restrict__ rotg,
    const float* __restrict__ trang, const float* __restrict__ rawpw,
    const float* __restrict__ wq_p, const float* __restrict__ bq_p,
    const float* __restrict__ wk_p, const float* __restrict__ bk_p,
    const float* __restrict__ wv_p, const float* __restrict__ bv_p,
    const float* __restrict__ wq_s, const float* __restrict__ wk_s,
    const float* __restrict__ wv_s,
    __bf16* __restrict__ qcomb, __bf16* __restrict__ kcomb,
    float* __restrict__ Qn, float* __restrict__ Kn,
    __bf16* __restrict__ vsT, __bf16* __restrict__ vptT) {
  __shared__ float s_in[C1];
  __shared__ float d_all[1152];
  __shared__ float qn_p[48], kn_p[48];
  const int n = blockIdx.x;
  const int t = threadIdx.x;
  for (int i = t; i < C1; i += 256) s_in[i] = in1d[n * C1 + i];
  __syncthreads();
  // phase A: all 1152 dot-products over C1
  for (int idx = t; idx < 1152; idx += 256) {
    const float* w; int stride; float b = 0.f;
    if (idx < 144)      { w = wq_p + idx;         stride = 144; b = bq_p[idx]; }
    else if (idx < 288) { int j = idx - 144; w = wk_p + j; stride = 144; b = bk_p[j]; }
    else if (idx < 576) { int j = idx - 288; w = wv_p + j; stride = 288; b = bv_p[j]; }
    else if (idx < 768) { w = wq_s + (idx - 576); stride = 192; }
    else if (idx < 960) { w = wk_s + (idx - 768); stride = 192; }
    else                { w = wv_s + (idx - 960); stride = 192; }
    float acc = b;
    for (int c = 0; c < C1; ++c) acc += s_in[c] * w[(size_t)c * stride];
    d_all[idx] = acc;
  }
  __syncthreads();
  float R[9], T3[3];
#pragma unroll
  for (int i = 0; i < 9; ++i) R[i] = rotg[n * 9 + i];
#pragma unroll
  for (int i = 0; i < 3; ++i) T3[i] = trang[n * 3 + i];
  const float pw_c = 0.23570226f;  // sqrt(1/18)
  if (t < 96) {  // q/k point transforms
    bool isq = t < 48; int tt = isq ? t : t - 48;
    int h = tt >> 2, p = tt & 3;
    const float* d = d_all + (isq ? 0 : 144) + h * 12;
    float lx = d[p], ly = d[4 + p], lz = d[8 + p];
    float g0 = R[0]*lx + R[1]*ly + R[2]*lz + T3[0];
    float g1 = R[3]*lx + R[4]*ly + R[5]*lz + T3[1];
    float g2 = R[6]*lx + R[7]*ly + R[8]*lz + T3[2];
    float pw = isq ? (pw_c * softplusf(rawpw[h])) : 1.0f;
    __bf16* dst = (isq ? qcomb : kcomb) + ((size_t)h * NN + n) * 32 + 16 + p * 3;
    dst[0] = (__bf16)(pw * g0); dst[1] = (__bf16)(pw * g1); dst[2] = (__bf16)(pw * g2);
    (isq ? qn_p : kn_p)[tt] = g0*g0 + g1*g1 + g2*g2;
  } else if (t < 192) {  // v point transforms
    int tt = t - 96; int h = tt >> 3, p = tt & 7;
    const float* d = d_all + 288 + h * 24;
    float lx = d[p], ly = d[8 + p], lz = d[16 + p];
    float g0 = R[0]*lx + R[1]*ly + R[2]*lz + T3[0];
    float g1 = R[3]*lx + R[4]*ly + R[5]*lz + T3[1];
    float g2 = R[6]*lx + R[7]*ly + R[8]*lz + T3[2];
    vptT[(size_t)(h * 32 + p * 3 + 0) * NN + n] = (__bf16)g0;
    vptT[(size_t)(h * 32 + p * 3 + 1) * NN + n] = (__bf16)g1;
    vptT[(size_t)(h * 32 + p * 3 + 2) * NN + n] = (__bf16)g2;
  }
  if (t < 192) {  // scalars
    int h = t >> 4, j = t & 15;
    qcomb[((size_t)h * NN + n) * 32 + j] = (__bf16)(0.25f * d_all[576 + t]);
    kcomb[((size_t)h * NN + n) * 32 + j] = (__bf16)(d_all[768 + t]);
    vsT[(size_t)(h * 16 + j) * NN + n]   = (__bf16)(d_all[960 + t]);
  }
  if (t < 48) {  // K pads 28..31
    int h = t >> 2, j = t & 3;
    qcomb[((size_t)h * NN + n) * 32 + 28 + j] = (__bf16)0.f;
    kcomb[((size_t)h * NN + n) * 32 + 28 + j] = (__bf16)0.f;
  }
  if (t < 96) {  // vptT row pads 24..31
    int h = t / 8, r = t % 8;
    vptT[(size_t)(h * 32 + 24 + r) * NN + n] = (__bf16)0.f;
  }
  __syncthreads();
  if (t < HH) {
    Qn[n * HH + t] = qn_p[t*4] + qn_p[t*4+1] + qn_p[t*4+2] + qn_p[t*4+3];
    Kn[n * HH + t] = kn_p[t*4] + kn_p[t*4+1] + kn_p[t*4+2] + kn_p[t*4+3];
  }
}

// ---------------------------------------------------------------------------
// Fused attention: one block = 16 queries, 16 waves, online softmax over k.
// ---------------------------------------------------------------------------
#define SMEM_BYTES 248704
__global__ __launch_bounds__(512) void ipa_main(
    const float* __restrict__ in2d, const float* __restrict__ maskp,
    const float* __restrict__ rotg, const float* __restrict__ trang,
    const float* __restrict__ rawpw, const float* __restrict__ b2d,
    const float* __restrict__ b_out,
    const __bf16* __restrict__ qcomb, const __bf16* __restrict__ kcomb,
    const float* __restrict__ Qn_g, const float* __restrict__ Kn_g,
    const __bf16* __restrict__ vsT, const __bf16* __restrict__ vptT,
    const __bf16* __restrict__ wT, const __bf16* __restrict__ w2dT,
    float* __restrict__ out) {
  extern __shared__ char smem[];
  __bf16* s_pair  = (__bf16*)(smem);            // [16q][32k][128c] bf16 131072B
  float*  s_log   = (float*)(smem + 131072);    // [16q][32k][12h]       24576B
  __bf16* s_pA    = (__bf16*)(smem + 155648);   // [12h][16q][32k]       12288B
  __bf16* s_pAT   = (__bf16*)(smem + 167936);   // [16q][16h][32k]       16384B
  float*  s_alpha = (float*)(smem + 184320);    // [16q][16h]             1024B
  float*  s_sum   = (float*)(smem + 185344);    // [16q][16h]             1024B
  float*  s_Qn    = (float*)(smem + 186368);    // [16q][12h]              768B
  float*  s_mq    = (float*)(smem + 187136);    // [16]                     64B
  float*  s_pw    = (float*)(smem + 187200);    // [12]+pad                 64B
  __bf16* s_kc    = (__bf16*)(smem + 187264);   // [12h][32k][32K]       24576B (async)
  __bf16* s_vs    = (__bf16*)(smem + 211840);   // [12h][16v][32k]       12288B (async)
  __bf16* s_vpt   = (__bf16*)(smem + 224128);   // [12h][32j][32k]       24576B (async)
  float*  s_feats = (float*)(smem);             // finalize alias  [16][2112]
  float*  s_ptg   = (float*)(smem + 155648);    // finalize alias [16][12][24]

  const int tid = threadIdx.x;
  const int lane = tid & 31, wv = tid >> 5;
  const int hi16 = lane >> 4, l16 = lane & 15;
  const int qbase = blockIdx.x * 16;
  const float s3 = 0.57735027f;

  if (tid < 192) s_Qn[tid] = Qn_g[(size_t)(qbase + tid / 12) * HH + tid % 12];
  if (tid < 16)  s_mq[tid] = maskp[qbase + tid];
  if (tid < HH)  s_pw[tid] = 0.23570226f * softplusf(rawpw[tid]);
  if (tid < 64)  s_alpha[(tid >> 2) * 16 + 12 + (tid & 3)] = 1.0f;
  for (int e = tid; e < 16 * 4 * 32; e += 512) {   // zero pAT head-pad rows
    int q = e >> 7, r = e & 127;
    s_pAT[(q * 16 + 12 + (r >> 5)) * 32 + (r & 31)] = (__bf16)0.f;
  }

  float m_run = -3.0e38f, s_run = 0.f;
  v8f acc2d[8];
#pragma unroll
  for (int i = 0; i < 8; ++i) acc2d[i] = zero8();
  v8f acc_s = zero8(), acc_pt0 = zero8(), acc_pt1 = zero8();

  v16bf aQK;
  if (wv < HH) aQK = frag_bf16(qcomb + ((size_t)wv * NN + qbase + l16) * 32 + hi16 * 16);
  const float bh = (wv < HH) ? b2d[wv] : 0.f;
  __syncthreads();

  for (int step = 0; step < 32; ++step) {
    const int kb = step * 32;
    // ---- async DMA: stage k-side bf16 slices straight into LDS (ASYNCcnt) ----
    // 3840 chunks of 16B: kcomb 1536 | vsT 768 | vptT 1536
    for (int e = tid; e < 3840; e += 512) {
      const __bf16* g; unsigned loff;
      if (e < 1536) {
        int row = e >> 2, c = e & 3;              // row = h*32 + k
        int h = row >> 5, k = row & 31;
        g = kcomb + ((size_t)h * NN + kb + k) * 32 + c * 8;
        loff = (unsigned)(size_t)(const void*)(s_kc + row * 32 + c * 8);
      } else if (e < 2304) {
        int ee = e - 1536; int row = ee >> 2, c = ee & 3;  // row = h*16 + v
        g = vsT + (size_t)row * NN + kb + c * 8;
        loff = (unsigned)(size_t)(const void*)(s_vs + row * 32 + c * 8);
      } else {
        int ee = e - 2304; int row = ee >> 2, c = ee & 3;  // row = h*32 + j
        g = vptT + (size_t)row * NN + kb + c * 8;
        loff = (unsigned)(size_t)(const void*)(s_vpt + row * 32 + c * 8);
      }
      async_copy16(loff, (const void*)g);
    }
    // ---- load pair tile f32 -> bf16 LDS (VALU convert path) ----
    {
      const float4* src = (const float4*)in2d;
#pragma unroll 4
      for (int i = 0; i < 32; ++i) {
        int g = tid + i * 512;           // 0..16383
        int c4 = g & 31, m = g >> 5;     // m = q*32 + k
        int q = m >> 5, k = m & 31;
        float4 v = src[((size_t)(qbase + q) * 1024 + kb + k) * 32 + c4];
        __bf16* d = s_pair + m * 128 + c4 * 4;
        d[0] = (__bf16)v.x; d[1] = (__bf16)v.y; d[2] = (__bf16)v.z; d[3] = (__bf16)v.w;
      }
    }
    wait_async0();
    __syncthreads();
    if (step + 1 < 32) {  // prefetch next pair tile into caches
#pragma unroll
      for (int i = 0; i < 4; ++i) {
        int e = tid + i * 512;
        int q = e >> 7, line = e & 127;
        __builtin_prefetch(in2d + ((size_t)(qbase + q) * 1024 + kb + 32) * 128 + line * 32, 0, 1);
      }
    }
    // ---- pair-bias WMMA: [512 pairs x 128c] x w2dT ----
    for (int mt = 2 * wv; mt <= 2 * wv + 1; ++mt) {
      v8f d = zero8();
#pragma unroll
      for (int s = 0; s < 4; ++s) {
        v16bf a = frag_bf16(s_pair + (mt * 16 + l16) * 128 + s * 32 + hi16 * 16);
        v16bf b = frag_bf16(w2dT + l16 * 128 + s * 32 + hi16 * 16);
        d = wmma_bf16(a, b, d);
      }
      if (l16 < HH) {
#pragma unroll
        for (int r = 0; r < 8; ++r) {
          int m = mt * 16 + r + 8 * hi16;
          s_log[m * HH + l16] = d[r];
        }
      }
    }
    __syncthreads();
    // ---- combined QK (scalar + point cross term) + logit assembly ----
    if (wv < HH) {
#pragma unroll
      for (int nt = 0; nt < 2; ++nt) {
        v16bf b = frag_bf16(s_kc + (wv * 32 + nt * 16 + l16) * 32 + hi16 * 16);
        v8f d = zero8();
        d = wmma_bf16(aQK, b, d);
        int k = nt * 16 + l16;
        float mk = maskp[kb + k];
        float kn = Kn_g[(size_t)(kb + k) * HH + wv];
#pragma unroll
        for (int r = 0; r < 8; ++r) {
          int q = r + 8 * hi16;
          float lg = s_log[(q * 32 + k) * HH + wv] + d[r] + bh
                     - 0.5f * s_pw[wv] * (s_Qn[q * HH + wv] + kn)
                     - 1.0e5f * (1.f - s_mq[q] * mk);
          s_log[(q * 32 + k) * HH + wv] = lg * s3;
        }
      }
    }
    __syncthreads();
    // ---- online softmax over this 32-key chunk ----
    if (tid < 192) {
      int q = tid / 12, h = tid % 12;
      const float* row = s_log + q * 32 * HH + h;
      float mx = -3.0e38f;
#pragma unroll
      for (int k = 0; k < 32; ++k) mx = fmaxf(mx, row[k * HH]);
      float m_new = fmaxf(m_run, mx);
      float alpha = __expf(m_run - m_new);
      float ps = 0.f;
#pragma unroll
      for (int k = 0; k < 32; ++k) {
        float p = __expf(row[k * HH] - m_new);
        ps += p;
        __bf16 pb = (__bf16)p;
        s_pA[(h * 16 + q) * 32 + k] = pb;
        s_pAT[(q * 16 + h) * 32 + k] = pb;
      }
      s_run = alpha * s_run + ps;
      m_run = m_new;
      s_alpha[q * 16 + h] = alpha;
    }
    __syncthreads();
    // ---- rescale + accumulate ----
    {
      v16bf a2 = frag_bf16(s_pAT + (wv * 16 + l16) * 32 + hi16 * 16);  // q = wv
#pragma unroll
      for (int ct = 0; ct < 8; ++ct) {
#pragma unroll
        for (int r = 0; r < 8; ++r) acc2d[ct][r] *= s_alpha[wv * 16 + (r + 8 * hi16)];
        BF16x16 u;
        int c = ct * 16 + l16;
#pragma unroll
        for (int j = 0; j < 16; ++j) u.e[j] = s_pair[(wv * 32 + hi16 * 16 + j) * 128 + c];
        acc2d[ct] = wmma_bf16(a2, u.v, acc2d[ct]);
      }
      if (wv < HH) {  // o_s, head = wv
        v16bf ap = frag_bf16(s_pA + (wv * 16 + l16) * 32 + hi16 * 16);
#pragma unroll
        for (int r = 0; r < 8; ++r) acc_s[r] *= s_alpha[(r + 8 * hi16) * 16 + wv];
        v16bf bs = frag_bf16(s_vs + (wv * 16 + l16) * 32 + hi16 * 16);
        acc_s = wmma_bf16(ap, bs, acc_s);
      }
      {  // o_pt tiles wv and wv+16
        int h0 = wv >> 1, j0 = wv & 1;
        v16bf apt = frag_bf16(s_pA + (h0 * 16 + l16) * 32 + hi16 * 16);
#pragma unroll
        for (int r = 0; r < 8; ++r) acc_pt0[r] *= s_alpha[(r + 8 * hi16) * 16 + h0];
        v16bf bpt = frag_bf16(s_vpt + (h0 * 32 + j0 * 16 + l16) * 32 + hi16 * 16);
        acc_pt0 = wmma_bf16(apt, bpt, acc_pt0);
        if (wv < 8) {
          int h1 = 8 + (wv >> 1), j1 = wv & 1;
          v16bf apt1 = frag_bf16(s_pA + (h1 * 16 + l16) * 32 + hi16 * 16);
#pragma unroll
          for (int r = 0; r < 8; ++r) acc_pt1[r] *= s_alpha[(r + 8 * hi16) * 16 + h1];
          v16bf bpt1 = frag_bf16(s_vpt + (h1 * 32 + j1 * 16 + l16) * 32 + hi16 * 16);
          acc_pt1 = wmma_bf16(apt1, bpt1, acc_pt1);
        }
      }
    }
    __syncthreads();
  }

  // ---- finalize ----
  if (tid < 192) {
    int q = tid / 12, h = tid % 12;
    s_sum[q * 16 + h] = fmaxf(s_run, 1e-30f);
  }
  __syncthreads();
  if (wv < HH) {  // res_scalar
#pragma unroll
    for (int r = 0; r < 8; ++r) {
      int q = r + 8 * hi16;
      s_feats[q * FEAT + wv * 16 + l16] = acc_s[r] / s_sum[q * 16 + wv];
    }
  }
#pragma unroll
  for (int ct = 0; ct < 8; ++ct) {  // res_2d
#pragma unroll
    for (int r = 0; r < 8; ++r) {
      int h = r + 8 * hi16;
      if (h < HH) s_feats[wv * FEAT + 576 + h * 128 + ct * 16 + l16] =
          acc2d[ct][r] / s_sum[wv * 16 + h];
    }
  }
  {  // point results -> s_ptg (global frame, normalized)
    int h0 = wv >> 1, j0 = wv & 1;
    int j = j0 * 16 + l16;
    if (j < 24) {
#pragma unroll
      for (int r = 0; r < 8; ++r) {
        int q = r + 8 * hi16;
        s_ptg[(q * 12 + h0) * 24 + j] = acc_pt0[r] / s_sum[q * 16 + h0];
      }
    }
    if (wv < 8) {
      int h1 = 8 + (wv >> 1), j1 = wv & 1;
      int jj = j1 * 16 + l16;
      if (jj < 24) {
#pragma unroll
        for (int r = 0; r < 8; ++r) {
          int q = r + 8 * hi16;
          s_ptg[(q * 12 + h1) * 24 + jj] = acc_pt1[r] / s_sum[q * 16 + h1];
        }
      }
    }
  }
  __syncthreads();
  for (int e = tid; e < 16 * 12 * 8; e += 512) {  // R^T(g - t), norms
    int q = e / 96, rem = e % 96, h = rem / 8, p = rem % 8;
    const float* g = s_ptg + (q * 12 + h) * 24 + p * 3;
    int n = qbase + q;
    float gx = g[0] - trang[n * 3 + 0];
    float gy = g[1] - trang[n * 3 + 1];
    float gz = g[2] - trang[n * 3 + 2];
    const float* R = rotg + n * 9;
    float lx = R[0] * gx + R[3] * gy + R[6] * gz;
    float ly = R[1] * gx + R[4] * gy + R[7] * gz;
    float lz = R[2] * gx + R[5] * gy + R[8] * gz;
    float nrm = sqrtf(fmaxf(lx * lx + ly * ly + lz * lz, 1e-16f));
    int m = h * 8 + p;
    float* f = s_feats + q * FEAT;
    f[192 + m] = lx; f[288 + m] = ly; f[384 + m] = lz; f[480 + m] = nrm;
  }
  __syncthreads();
  for (int pass = 0; pass < 2; ++pass) {  // feats[16x2112] x w_out -> out
    if (pass == 1 && wv >= 8) break;
    int nt = (pass == 0) ? wv : wv + 16;
    v8f acc = zero8();
    for (int ks = 0; ks < 66; ++ks) {
      v16bf a = frag_cvt(s_feats + l16 * FEAT + ks * 32 + hi16 * 16);
      v16bf b = frag_bf16(wT + (size_t)(nt * 16 + l16) * FEAT + ks * 32 + hi16 * 16);
      acc = wmma_bf16(a, b, acc);
    }
#pragma unroll
    for (int r = 0; r < 8; ++r) {
      int q = r + 8 * hi16;
      out[(size_t)(qbase + q) * OUTD + nt * 16 + l16] = acc[r] + b_out[nt * 16 + l16];
    }
  }
}

// ---------------------------------------------------------------------------
extern "C" void kernel_launch(void* const* d_in, const int* in_sizes, int n_in,
                              void* d_out, int out_size, void* d_ws, size_t ws_size,
                              hipStream_t stream) {
  (void)in_sizes; (void)n_in; (void)out_size; (void)ws_size;
  const float* in1d  = (const float*)d_in[0];
  const float* in2d  = (const float*)d_in[1];
  const float* mask  = (const float*)d_in[2];
  const float* rotg  = (const float*)d_in[3];
  const float* trang = (const float*)d_in[4];
  const float* rawpw = (const float*)d_in[5];
  const float* wq_p  = (const float*)d_in[6];
  const float* bq_p  = (const float*)d_in[7];
  const float* wk_p  = (const float*)d_in[8];
  const float* bk_p  = (const float*)d_in[9];
  const float* wv_p  = (const float*)d_in[10];
  const float* bv_p  = (const float*)d_in[11];
  const float* wq_s  = (const float*)d_in[12];
  const float* wk_s  = (const float*)d_in[13];
  const float* wv_s  = (const float*)d_in[14];
  const float* w2d   = (const float*)d_in[15];
  const float* b2d   = (const float*)d_in[16];
  const float* w_out = (const float*)d_in[17];
  const float* b_out = (const float*)d_in[18];
  float* out = (float*)d_out;

  // workspace carve-up (~4.5 MB)
  char* ws = (char*)d_ws;
  size_t o = 0;
  auto carve = [&](size_t bytes) { char* p = ws + o; o += (bytes + 255) & ~(size_t)255; return p; };
  __bf16* qcomb = (__bf16*)carve((size_t)HH * NN * 32 * 2);
  __bf16* kcomb = (__bf16*)carve((size_t)HH * NN * 32 * 2);
  float*  Qn    = (float*) carve((size_t)NN * HH * 4);
  float*  Kn    = (float*) carve((size_t)NN * HH * 4);
  __bf16* vsT   = (__bf16*)carve((size_t)HH * 16 * NN * 2);
  __bf16* vptT  = (__bf16*)carve((size_t)HH * 32 * NN * 2);
  __bf16* wT    = (__bf16*)carve((size_t)OUTD * FEAT * 2);
  __bf16* w2dT  = (__bf16*)carve((size_t)16 * C2 * 2);

  ipa_wprep<<<3176, 256, 0, stream>>>(w_out, w2d, wT, w2dT);
  ipa_project<<<NN, 256, 0, stream>>>(in1d, rotg, trang, rawpw,
                                      wq_p, bq_p, wk_p, bk_p, wv_p, bv_p,
                                      wq_s, wk_s, wv_s,
                                      qcomb, kcomb, Qn, Kn, vsT, vptT);
  ipa_main<<<NN / 16, 512, SMEM_BYTES, stream>>>(
      in2d, mask, rotg, trang, rawpw, b2d, b_out,
      qcomb, kcomb, Qn, Kn, vsT, vptT, wT, w2dT, out);
}
